// DEC_21732534517794
// MI455X (gfx1250) — compile-verified, matrix operand order checked by reading, not run on hardware
//
#include <hip/hip_runtime.h>
#include <hip/hip_bf16.h>

typedef __attribute__((ext_vector_type(16))) _Float16 v16h;
typedef __attribute__((ext_vector_type(8)))  float    v8f;
typedef __attribute__((ext_vector_type(4)))  float    v4f;

#define HDIM   128
#define LSTEPS 100
#define BT     16
#define NBT    8
#define G3     384   // 3*H

struct HalfPair { v4f x, y; };

// Load one 16x32 f16 WMMA fragment (A layout; B uses same mapping on W rows).
// lane<16: rows r=lane, K[kb+0..7] and K[kb+16..23]; lane>=16: K[kb+8..15], K[kb+24..31]
__device__ __forceinline__ v16h ldfrag(const _Float16* base, int ld, int lane, int kb) {
  int r  = lane & 15;
  int kc = (lane >> 4) << 3;
  const _Float16* p = base + r * ld + kb + kc;
  HalfPair t;
  t.x = *(const v4f*)(p);
  t.y = *(const v4f*)(p + 16);
  return __builtin_bit_cast(v16h, t);
}

// Fast transcendentals: single v_exp_f32 + v_rcp_f32, no IEEE divide expansion.
__device__ __forceinline__ float fast_sigm(float x) {
  return __builtin_amdgcn_rcpf(1.0f + __expf(-x));
}
__device__ __forceinline__ float fast_tanh(float x) {
  // tanh(x) = 1 - 2/(exp(2x)+1); saturates correctly at +-1.
  float e = __expf(2.0f * x);
  return 1.0f - 2.0f * __builtin_amdgcn_rcpf(e + 1.0f);
}

// ---------------------------------------------------------------------------
// Prologue: convert / repack weights to f16
// ---------------------------------------------------------------------------
__global__ void convert_weights(const float* __restrict__ Whh0,
                                const float* __restrict__ Wih1,
                                const float* __restrict__ Whh1,
                                const float* __restrict__ Wfc,
                                const float* __restrict__ Wattn,
                                _Float16* __restrict__ whh0h,
                                _Float16* __restrict__ wih1h,
                                _Float16* __restrict__ whh1h,
                                _Float16* __restrict__ wfcC,
                                _Float16* __restrict__ wfcH,
                                _Float16* __restrict__ waS,
                                _Float16* __restrict__ waK) {
  int idx = blockIdx.x * blockDim.x + threadIdx.x;
  const int NW = 2 * G3 * HDIM;
  if (idx < NW) {
    whh0h[idx] = (_Float16)Whh0[idx];
    wih1h[idx] = (_Float16)Wih1[idx];
    whh1h[idx] = (_Float16)Whh1[idx];
  }
  if (idx < HDIM * HDIM) {
    int g = idx >> 7, h = idx & 127;
    wfcC[idx] = (_Float16)Wfc[g * 256 + h];
    wfcH[idx] = (_Float16)Wfc[g * 256 + 128 + h];
    waS[idx]  = (_Float16)Wattn[g * 256 + h];
    waK[idx]  = (_Float16)Wattn[g * 256 + 128 + h];
  }
}

// ---------------------------------------------------------------------------
// Persistent decoder kernel: one WG per (batch-tile, decoder k), 100 steps.
// ---------------------------------------------------------------------------
__global__ __launch_bounds__(256)
void decoder_kernel(const float* __restrict__ recv,
                    const float* __restrict__ Wih0,
                    const float* __restrict__ bih0,
                    const float* __restrict__ bhh0,
                    const float* __restrict__ bih1,
                    const float* __restrict__ bhh1,
                    const float* __restrict__ bfc,
                    const float* __restrict__ Wv,
                    const float* __restrict__ Wout,
                    const _Float16* __restrict__ whh0h,
                    const _Float16* __restrict__ wih1h,
                    const _Float16* __restrict__ whh1h,
                    const _Float16* __restrict__ wfcC16,
                    const _Float16* __restrict__ wfcH16,
                    const _Float16* __restrict__ waS16,
                    const _Float16* __restrict__ waK16,
                    _Float16* __restrict__ keysG,
                    _Float16* __restrict__ kprojG,
                    float* __restrict__ pout) {
  const int tid   = threadIdx.x;
  const int lane  = tid & 31;
  const int wave  = tid >> 5;
  const int bt    = blockIdx.x;   // batch tile 0..7
  const int k     = blockIdx.y;   // decoder stack 0..1
  const int bbase = bt * BT;

  const _Float16* whh0k = whh0h + k * G3 * HDIM;
  const _Float16* wih1k = wih1h + k * G3 * HDIM;
  const _Float16* whh1k = whh1h + k * G3 * HDIM;

  _Float16* myKeys  = keysG  + (size_t)(k * NBT + bt) * (LSTEPS + 1) * (BT * HDIM);
  _Float16* myKproj = kprojG + (size_t)(k * NBT + bt) * (LSTEPS + 1) * (BT * HDIM);

  // Persistent state (f32 master + f16 WMMA operand copies)
  __shared__ __align__(16) _Float16 h0_16[BT * HDIM],  h1_16[BT * HDIM];
  __shared__ __align__(16) _Float16 h0n_16[BT * HDIM], h1n_16[BT * HDIM];
  __shared__ __align__(16) _Float16 c11_16[BT * HDIM], c12_16[BT * HDIM];
  __shared__ __align__(16) float    h0_32[BT * HDIM],  h1_32[BT * HDIM];
  __shared__ __align__(16) float    h0n_32[BT * HDIM], h1n_32[BT * HDIM];
  __shared__ __align__(16) float    c11_32[BT * HDIM], c12_32[BT * HDIM];
  // Small constants cached in LDS (inner-loop operands)
  __shared__ __align__(16) float    WvL[HDIM], bfcL[HDIM], woutL[HDIM];
  __shared__ __align__(16) float    bih0L[G3], bhh0L[G3], bih1L[G3], bhh1L[G3];
  __shared__ __align__(16) float    wih0L[G3 * 3];
  // Aliased scratch region: gi/gh (gates) vs attention proj/score buffers
  __shared__ __align__(16) float    regA[2 * BT * G3];

  float* gi     = regA;                 // [16][384]
  float* gh     = regA + BT * G3;       // [16][384]
  float* sproj0 = regA;                 // [16][128]
  float* sproj1 = regA + 2048;
  float* kpnew  = regA + 4096;
  float* kph1   = regA + 6144;
  float* sc     = regA + 8192;          // [16][104] scores

  for (int idx = tid; idx < BT * HDIM; idx += 256) {
    h0_32[idx] = 0.f; h1_32[idx] = 0.f;
    h0_16[idx] = (_Float16)0.f; h1_16[idx] = (_Float16)0.f;
  }
  for (int idx = tid; idx < G3; idx += 256) {
    bih0L[idx] = bih0[k * G3 + idx];
    bhh0L[idx] = bhh0[k * G3 + idx];
    bih1L[idx] = bih1[k * G3 + idx];
    bhh1L[idx] = bhh1[k * G3 + idx];
  }
  for (int idx = tid; idx < HDIM; idx += 256) {
    WvL[idx]   = Wv[idx];
    bfcL[idx]  = bfc[idx];
    woutL[idx] = Wout[k * HDIM + idx];
  }
  for (int idx = tid; idx < G3 * 3; idx += 256) wih0L[idx] = Wih0[k * G3 * 3 + idx];
  __syncthreads();

  for (int i = 0; i < LSTEPS; ++i) {
    // ---- P0: gi0 = x @ Wih0.T + bih0 (IN=3, scalar) ----
    for (int idx = tid; idx < BT * G3; idx += 256) {
      int b = idx / G3, g = idx - b * G3;
      const float* xr = recv + ((size_t)(bbase + b) * LSTEPS + i) * 3;
      const float* w  = wih0L + g * 3;
      gi[idx] = bih0L[g] + xr[0] * w[0] + xr[1] * w[1] + xr[2] * w[2];
    }
    // ---- P1: gh0 = h0 @ Whh0.T  (24 tiles, WMMA) ----
    for (int tile = wave; tile < 24; tile += 8) {
      const _Float16* Bw = whh0k + tile * 16 * HDIM;
      v8f c = {};
#pragma unroll
      for (int kb = 0; kb < HDIM; kb += 32) {
        v16h av = ldfrag(h0_16, HDIM, lane, kb);
        v16h bv = ldfrag(Bw, HDIM, lane, kb);
        c = __builtin_amdgcn_wmma_f32_16x16x32_f16(false, av, false, bv, (short)0, c, false, false);
      }
      int mb = (lane >> 4) * 8, n = lane & 15;
#pragma unroll
      for (int j = 0; j < 8; ++j) gh[(mb + j) * G3 + tile * 16 + n] = c[j];
    }
    __syncthreads();
    // ---- P2: GRU cell0 gate combine -> h0n; append key ----
    for (int idx = tid; idx < BT * HDIM; idx += 256) {
      int b = idx >> 7, h2 = idx & 127;
      float ir  = gi[b * G3 + h2];
      float iz  = gi[b * G3 + 128 + h2];
      float in_ = gi[b * G3 + 256 + h2];
      float hr  = gh[b * G3 + h2]       + bhh0L[h2];
      float hz  = gh[b * G3 + 128 + h2] + bhh0L[128 + h2];
      float hn  = gh[b * G3 + 256 + h2] + bhh0L[256 + h2];
      float r = fast_sigm(ir + hr);
      float z = fast_sigm(iz + hz);
      float n = fast_tanh(in_ + r * hn);
      float v = (1.f - z) * n + z * h0_32[idx];
      h0n_32[idx] = v;
      h0n_16[idx] = (_Float16)v;
      myKeys[(size_t)i * (BT * HDIM) + idx] = (_Float16)v;
    }
    __syncthreads();
    // ---- P3: gi1 = h0n @ Wih1.T ; gh1 = h1 @ Whh1.T (48 tiles, WMMA) ----
    for (int tile = wave; tile < 48; tile += 8) {
      bool isGh = tile >= 24;
      int  t2   = isGh ? tile - 24 : tile;
      const _Float16* A  = isGh ? h1_16 : h0n_16;
      const _Float16* Bw = (isGh ? whh1k : wih1k) + t2 * 16 * HDIM;
      v8f c = {};
#pragma unroll
      for (int kb = 0; kb < HDIM; kb += 32) {
        v16h av = ldfrag(A, HDIM, lane, kb);
        v16h bv = ldfrag(Bw, HDIM, lane, kb);
        c = __builtin_amdgcn_wmma_f32_16x16x32_f16(false, av, false, bv, (short)0, c, false, false);
      }
      float* out = isGh ? gh : gi;
      int mb = (lane >> 4) * 8, n = lane & 15;
#pragma unroll
      for (int j = 0; j < 8; ++j) out[(mb + j) * G3 + t2 * 16 + n] = c[j];
    }
    __syncthreads();
    // ---- P4: GRU cell1 gate combine -> h1n ----
    for (int idx = tid; idx < BT * HDIM; idx += 256) {
      int b = idx >> 7, h2 = idx & 127;
      float ir  = gi[b * G3 + h2]       + bih1L[h2];
      float iz  = gi[b * G3 + 128 + h2] + bih1L[128 + h2];
      float in_ = gi[b * G3 + 256 + h2] + bih1L[256 + h2];
      float hr  = gh[b * G3 + h2]       + bhh1L[h2];
      float hz  = gh[b * G3 + 128 + h2] + bhh1L[128 + h2];
      float hn  = gh[b * G3 + 256 + h2] + bhh1L[256 + h2];
      float r = fast_sigm(ir + hr);
      float z = fast_sigm(iz + hz);
      float n = fast_tanh(in_ + r * hn);
      float v = (1.f - z) * n + z * h1_32[idx];
      h1n_32[idx] = v;
      h1n_16[idx] = (_Float16)v;
    }
    __syncthreads();
    // ---- P5: projections sproj0, kpnew, sproj1, kph1 (32 tiles, WMMA) ----
    for (int tile = wave; tile < 32; tile += 8) {
      int which = tile >> 3, nt = tile & 7;
      const _Float16* A  = (which < 2) ? h0n_16 : h1n_16;
      const _Float16* Bw = (((which & 1) == 0) ? waS16 : waK16) + nt * 16 * HDIM;
      v8f c = {};
#pragma unroll
      for (int kb = 0; kb < HDIM; kb += 32) {
        v16h av = ldfrag(A, HDIM, lane, kb);
        v16h bv = ldfrag(Bw, HDIM, lane, kb);
        c = __builtin_amdgcn_wmma_f32_16x16x32_f16(false, av, false, bv, (short)0, c, false, false);
      }
      float* out = (which == 0) ? sproj0 : (which == 1) ? kpnew : (which == 2) ? sproj1 : kph1;
      int mb = (lane >> 4) * 8, n = lane & 15;
#pragma unroll
      for (int j = 0; j < 8; ++j) out[(mb + j) * HDIM + nt * 16 + n] = c[j];
    }
    __syncthreads();
    for (int idx = tid; idx < BT * HDIM; idx += 256)
      myKproj[(size_t)i * (BT * HDIM) + idx] = (_Float16)kpnew[idx];
    __threadfence();
    __syncthreads();

    // ---- attention c11: keys = buf[0..i] (slot i in LDS) ----
    {
      int nk = i + 1;
      for (int idx = tid; idx < nk * BT; idx += 256) {
        int b = idx & 15, t = idx >> 4;
        const float* sp = sproj0 + b * HDIM;
        float s = 0.f;
        if (t == i) {
          const float* kp = kpnew + b * HDIM;
          for (int h2 = 0; h2 < HDIM; ++h2) s += fast_tanh(sp[h2] + kp[h2]) * WvL[h2];
        } else {
          const _Float16* kp = myKproj + (size_t)t * (BT * HDIM) + b * HDIM;
          for (int h2 = 0; h2 < HDIM; ++h2) s += fast_tanh(sp[h2] + (float)kp[h2]) * WvL[h2];
        }
        sc[b * 104 + t] = s;
      }
      __syncthreads();
      {  // parallel softmax: 16 lanes per batch row, shfl_xor reductions
        int b = tid >> 4, sub = tid & 15;
        float mx = -1e30f;
        for (int t = sub; t < nk; t += 16) mx = fmaxf(mx, sc[b * 104 + t]);
        for (int m = 8; m >= 1; m >>= 1) mx = fmaxf(mx, __shfl_xor(mx, m, 16));
        float sum = 0.f;
        for (int t = sub; t < nk; t += 16) { float e = __expf(sc[b * 104 + t] - mx); sc[b * 104 + t] = e; sum += e; }
        for (int m = 8; m >= 1; m >>= 1) sum += __shfl_xor(sum, m, 16);
        float inv = __builtin_amdgcn_rcpf(sum);
        for (int t = sub; t < nk; t += 16) sc[b * 104 + t] *= inv;
      }
      __syncthreads();
      for (int idx = tid; idx < BT * HDIM; idx += 256) {
        int b = idx >> 7, h2 = idx & 127;
        float acc = 0.f;
        for (int t = 0; t < nk; ++t) {
          float kv = (t == i) ? h0n_32[idx]
                              : (float)myKeys[(size_t)t * (BT * HDIM) + b * HDIM + h2];
          acc += sc[b * 104 + t] * kv;
        }
        c11_32[idx] = acc; c11_16[idx] = (_Float16)acc;
      }
      __syncthreads();
    }
    // ---- attention c12: keys = buf[lo..i] + h1n at slot i+1 ----
    {
      int lo = (i == 0) ? 1 : 0;
      int nk = i + 2 - lo;
      for (int idx = tid; idx < nk * BT; idx += 256) {
        int b = idx & 15, t = lo + (idx >> 4);
        const float* sp = sproj1 + b * HDIM;
        float s = 0.f;
        if (t == i + 1) {
          const float* kp = kph1 + b * HDIM;
          for (int h2 = 0; h2 < HDIM; ++h2) s += fast_tanh(sp[h2] + kp[h2]) * WvL[h2];
        } else if (t == i) {
          const float* kp = kpnew + b * HDIM;
          for (int h2 = 0; h2 < HDIM; ++h2) s += fast_tanh(sp[h2] + kp[h2]) * WvL[h2];
        } else {
          const _Float16* kp = myKproj + (size_t)t * (BT * HDIM) + b * HDIM;
          for (int h2 = 0; h2 < HDIM; ++h2) s += fast_tanh(sp[h2] + (float)kp[h2]) * WvL[h2];
        }
        sc[b * 104 + (t - lo)] = s;
      }
      __syncthreads();
      {
        int b = tid >> 4, sub = tid & 15;
        float mx = -1e30f;
        for (int t = sub; t < nk; t += 16) mx = fmaxf(mx, sc[b * 104 + t]);
        for (int m = 8; m >= 1; m >>= 1) mx = fmaxf(mx, __shfl_xor(mx, m, 16));
        float sum = 0.f;
        for (int t = sub; t < nk; t += 16) { float e = __expf(sc[b * 104 + t] - mx); sc[b * 104 + t] = e; sum += e; }
        for (int m = 8; m >= 1; m >>= 1) sum += __shfl_xor(sum, m, 16);
        float inv = __builtin_amdgcn_rcpf(sum);
        for (int t = sub; t < nk; t += 16) sc[b * 104 + t] *= inv;
      }
      __syncthreads();
      for (int idx = tid; idx < BT * HDIM; idx += 256) {
        int b = idx >> 7, h2 = idx & 127;
        float acc = 0.f;
        for (int tt = 0; tt < nk; ++tt) {
          int t = lo + tt;
          float kv = (t == i + 1) ? h1n_32[idx]
                   : (t == i)     ? h0n_32[idx]
                                  : (float)myKeys[(size_t)t * (BT * HDIM) + b * HDIM + h2];
          acc += sc[b * 104 + tt] * kv;
        }
        c12_32[idx] = acc; c12_16[idx] = (_Float16)acc;
      }
      __syncthreads();
    }
    // ---- P11: output partial p[b][i] = h1n . Wout[k*H ..] (16 lanes/row) ----
    {
      int b = tid >> 4, sub = tid & 15;
      float acc = 0.f;
      for (int h2 = sub; h2 < HDIM; h2 += 16) acc += h1n_32[b * HDIM + h2] * woutL[h2];
      for (int m = 8; m >= 1; m >>= 1) acc += __shfl_xor(acc, m, 16);
      if (sub == 0)
        pout[(size_t)k * 12800 + (size_t)(bbase + b) * LSTEPS + i] = acc;
    }
    // ---- P10: h_new = c @ WfcC.T + hn @ WfcH.T + bfc (16 tiles, 8 WMMA each) ----
    for (int job = wave; job < 16; job += 8) {
      int sel = job >> 3, nt = job & 7;
      const _Float16* Ac = sel ? c12_16 : c11_16;
      const _Float16* Ah = sel ? h1n_16 : h0n_16;
      const _Float16* Bc = wfcC16 + nt * 16 * HDIM;
      const _Float16* Bh = wfcH16 + nt * 16 * HDIM;
      v8f c = {};
#pragma unroll
      for (int kb = 0; kb < HDIM; kb += 32) {
        v16h av = ldfrag(Ac, HDIM, lane, kb);
        v16h bv = ldfrag(Bc, HDIM, lane, kb);
        c = __builtin_amdgcn_wmma_f32_16x16x32_f16(false, av, false, bv, (short)0, c, false, false);
      }
#pragma unroll
      for (int kb = 0; kb < HDIM; kb += 32) {
        v16h av = ldfrag(Ah, HDIM, lane, kb);
        v16h bv = ldfrag(Bh, HDIM, lane, kb);
        c = __builtin_amdgcn_wmma_f32_16x16x32_f16(false, av, false, bv, (short)0, c, false, false);
      }
      float*    H32 = sel ? h1_32 : h0_32;
      _Float16* H16 = sel ? h1_16 : h0_16;
      int mb = (lane >> 4) * 8, n = lane & 15;
#pragma unroll
      for (int j = 0; j < 8; ++j) {
        float v = c[j] + bfcL[nt * 16 + n];
        int o = (mb + j) * HDIM + nt * 16 + n;
        H32[o] = v; H16[o] = (_Float16)v;
      }
    }
    __syncthreads();
  }
}

// ---------------------------------------------------------------------------
// Epilogue: out[b][l] = sigmoid(p0[b][l] + p1[b][min(l+1,L-1)] + bout)
// ---------------------------------------------------------------------------
__global__ void finalize_kernel(const float* __restrict__ pout,
                                const float* __restrict__ bout,
                                float* __restrict__ out) {
  int idx = blockIdx.x * blockDim.x + threadIdx.x;
  if (idx < 128 * LSTEPS) {
    int b = idx / LSTEPS, l = idx - b * LSTEPS;
    int l2 = (l + 1 < LSTEPS) ? l + 1 : LSTEPS - 1;
    float v = pout[(size_t)b * LSTEPS + l] + pout[12800 + (size_t)b * LSTEPS + l2] + bout[0];
    out[idx] = __builtin_amdgcn_rcpf(1.0f + __expf(-v));
  }
}

extern "C" void kernel_launch(void* const* d_in, const int* in_sizes, int n_in,
                              void* d_out, int out_size, void* d_ws, size_t ws_size,
                              hipStream_t stream) {
  const float* recv  = (const float*)d_in[0];
  const float* Wih0  = (const float*)d_in[1];
  const float* Whh0  = (const float*)d_in[2];
  const float* bih0  = (const float*)d_in[3];
  const float* bhh0  = (const float*)d_in[4];
  const float* Wih1  = (const float*)d_in[5];
  const float* Whh1  = (const float*)d_in[6];
  const float* bih1  = (const float*)d_in[7];
  const float* bhh1  = (const float*)d_in[8];
  const float* Wfc   = (const float*)d_in[9];
  const float* bfc   = (const float*)d_in[10];
  const float* Wattn = (const float*)d_in[11];
  const float* Wv    = (const float*)d_in[12];
  const float* Wout  = (const float*)d_in[13];
  const float* bout  = (const float*)d_in[14];

  char* ws = (char*)d_ws;
  size_t off = 0;
  auto take = [&](size_t bytes) -> char* {
    char* p = ws + off;
    off += (bytes + 255) & ~(size_t)255;
    return p;
  };
  const size_t NW   = (size_t)2 * G3 * HDIM;                  // 98304
  const size_t HH   = (size_t)HDIM * HDIM;                    // 16384
  const size_t KBUF = (size_t)16 * (LSTEPS + 1) * BT * HDIM;  // all (k,bt) slots

  _Float16* whh0h  = (_Float16*)take(NW * 2);
  _Float16* wih1h  = (_Float16*)take(NW * 2);
  _Float16* whh1h  = (_Float16*)take(NW * 2);
  _Float16* wfcC16 = (_Float16*)take(HH * 2);
  _Float16* wfcH16 = (_Float16*)take(HH * 2);
  _Float16* waS16  = (_Float16*)take(HH * 2);
  _Float16* waK16  = (_Float16*)take(HH * 2);
  _Float16* keysG  = (_Float16*)take(KBUF * 2);
  _Float16* kprojG = (_Float16*)take(KBUF * 2);
  float*    pout   = (float*)take((size_t)2 * 128 * LSTEPS * 4);
  (void)ws_size; (void)in_sizes; (void)n_in; (void)out_size;

  convert_weights<<<(NW + 255) / 256, 256, 0, stream>>>(
      Whh0, Wih1, Whh1, Wfc, Wattn, whh0h, wih1h, whh1h, wfcC16, wfcH16, waS16, waK16);

  decoder_kernel<<<dim3(NBT, 2), 256, 0, stream>>>(
      recv, Wih0, bih0, bhh0, bih1, bhh1, bfc, Wv, Wout,
      whh0h, wih1h, whh1h, wfcC16, wfcH16, waS16, waK16,
      keysG, kprojG, pout);

  finalize_kernel<<<(128 * LSTEPS + 255) / 256, 256, 0, stream>>>(pout, bout, (float*)d_out);
}
